// MultiHeadAttention_41549513621525
// MI455X (gfx1250) — matrix operand order repaired and checked by reading, not compiled, not run on hardware
//
#include <hip/hip_runtime.h>
#include <hip/hip_bf16.h>

typedef __attribute__((ext_vector_type(8)))  float   v8f;
typedef __attribute__((ext_vector_type(16))) __bf16  v16bf;
typedef __attribute__((ext_vector_type(8)))  __bf16  v8bf;

#define BLOCK_M 128
#define BLOCK_N 64
#define TILE_K  32
#define LDS_STRIDE 40   // bf16 elems per LDS row: 80 B, 16B-aligned rows, breaks bank cycles

// ---- CDNA5 async global->LDS copy (ASYNCcnt path) ----
__device__ __forceinline__ void async_copy_b128(const void* gsrc, void* ldsdst) {
  unsigned lds = (unsigned)(unsigned long long)ldsdst;   // low 32 bits = LDS offset
  asm volatile("global_load_async_to_lds_b128 %0, %1, off"
               :: "v"(lds), "v"((unsigned long long)gsrc)
               : "memory");
}
__device__ __forceinline__ void wait_async() {
  asm volatile("s_wait_asynccnt 0x0" ::: "memory");
}

// Assemble a 16-element bf16 WMMA fragment from two contiguous 16-byte LDS reads.
__device__ __forceinline__ v16bf make_frag(const __bf16* p0, const __bf16* p1) {
  v8bf lo = *(const v8bf*)p0;
  v8bf hi = *(const v8bf*)p1;
  v16bf r;
#pragma unroll
  for (int i = 0; i < 8; ++i) { r[i] = lo[i]; r[i + 8] = hi[i]; }
  return r;
}

__device__ __forceinline__ v8bf cvt8(float4 f0, float4 f1) {
  v8bf a8;
  a8[0] = (__bf16)f0.x; a8[1] = (__bf16)f0.y; a8[2] = (__bf16)f0.z; a8[3] = (__bf16)f0.w;
  a8[4] = (__bf16)f1.x; a8[5] = (__bf16)f1.y; a8[6] = (__bf16)f1.z; a8[7] = (__bf16)f1.w;
  return a8;
}

// Generic C = A * Bn^T (+bias) with bf16 WMMA, double-buffered LDS pipeline.
//   A : row-major [M x K], fp32 (cvt during staging) or bf16 (async-to-LDS)
//   Bn: row-major [N x K] (already transposed-B layout; W matrices are naturally
//       [N x K] for X @ W^T), fp32 or bf16 (async-to-LDS)
//   OUT_MODE 0: bf16 row-major   1: bf16 vT scatter [b][h][d][s]   2: fp32 row-major
// blockIdx.z = b*16 + h for batched-head GEMMs; per-z element offsets via s?b/s?h.
template <int A_IS_F32, int B_IS_F32, int OUT_MODE, int HAS_BIAS>
__global__ void __launch_bounds__(256)
gemm_wmma_bf16(const void* __restrict__ Abase, long long lda,
               const void* __restrict__ Bbase, long long ldb,
               void* __restrict__ Obase, long long ldo,
               const float* __restrict__ bias,
               int M, int N, int K,
               long long sAb, long long sAh,
               long long sBb, long long sBh,
               long long sOb, long long sOh) {
  const int zb = blockIdx.z >> 4;   // H == 16
  const int zh = blockIdx.z & 15;
  const long long offA = (long long)zb * sAb + (long long)zh * sAh;
  const long long offB = (long long)zb * sBb + (long long)zh * sBh;
  const long long offO = (long long)zb * sOb + (long long)zh * sOh;

  const int mBlock = blockIdx.y * BLOCK_M;
  const int nBlock = blockIdx.x * BLOCK_N;

  __shared__ __bf16 At[2][BLOCK_M][LDS_STRIDE];   // [buf][m][k]
  __shared__ __bf16 Bt[2][BLOCK_N][LDS_STRIDE];   // [buf][n][k]

  const int tid  = threadIdx.x;
  const int lane = tid & 31;
  const int wv   = tid >> 5;          // 8 wave32
  const int hlf  = lane >> 4;         // lane half per ISA 7.12.2 layouts
  const int l16  = lane & 15;
  const int mOff = (wv & 3) * 32;     // wave tile: 32 x 32
  const int nOff = (wv >> 2) * 32;

  v8f acc00 = {}, acc01 = {}, acc10 = {}, acc11 = {};

  const int ar = tid >> 2, ac = (tid & 3) * 8;   // A staging: 2 x (64 rows x 32 cols)
  const int br = tid >> 2, bc = (tid & 3) * 8;   // B staging: 64 rows x 32 cols

  const float*  Af = (const float*)Abase  + offA;
  const __bf16* Ab = (const __bf16*)Abase + offA;
  const float*  Bf = (const float*)Bbase  + offB;
  const __bf16* Bb = (const __bf16*)Bbase + offB;
  const long long aRow = (long long)(mBlock + ar) * lda + ac;
  const long long bRow = (long long)(nBlock + br) * ldb + bc;

  // ---------------- prologue: stage k0 = 0 into buffer 0 ----------------
  if (A_IS_F32) {
    const float* s0 = Af + aRow;
    const float* s1 = s0 + 64 * lda;
    *(v8bf*)&At[0][ar][ac]      = cvt8(*(const float4*)s0, *(const float4*)(s0 + 4));
    *(v8bf*)&At[0][ar + 64][ac] = cvt8(*(const float4*)s1, *(const float4*)(s1 + 4));
  } else {
    const __bf16* s0 = Ab + aRow;
    async_copy_b128(s0,            &At[0][ar][ac]);
    async_copy_b128(s0 + 64 * lda, &At[0][ar + 64][ac]);
  }
  if (B_IS_F32) {
    const float* s = Bf + bRow;
    *(v8bf*)&Bt[0][br][bc] = cvt8(*(const float4*)s, *(const float4*)(s + 4));
  } else {
    async_copy_b128(Bb + bRow, &Bt[0][br][bc]);
  }
  if (!A_IS_F32 || !B_IS_F32) wait_async();
  __syncthreads();

  // ---------------- main pipelined loop ----------------
  int cur = 0;
  for (int k0 = 0; k0 < K; k0 += TILE_K) {
    const int  nxt     = cur ^ 1;
    const bool hasNext = (k0 + TILE_K) < K;
    const int  kn      = k0 + TILE_K;

    // -- issue next tile's global traffic before compute --
    float4 fa0, fa1, fa2, fa3, fb0, fb1;
    if (hasNext) {
      if (A_IS_F32) {
        const float* s0 = Af + aRow + kn;
        const float* s1 = s0 + 64 * lda;
        fa0 = *(const float4*)s0; fa1 = *(const float4*)(s0 + 4);
        fa2 = *(const float4*)s1; fa3 = *(const float4*)(s1 + 4);
        if (kn + TILE_K < K) {
          __builtin_prefetch(s0 + TILE_K, 0, 1);   // global_prefetch_b8
          __builtin_prefetch(s1 + TILE_K, 0, 1);
        }
      } else {
        const __bf16* s0 = Ab + aRow + kn;
        async_copy_b128(s0,            &At[nxt][ar][ac]);
        async_copy_b128(s0 + 64 * lda, &At[nxt][ar + 64][ac]);
      }
      if (B_IS_F32) {
        const float* s = Bf + bRow + kn;
        fb0 = *(const float4*)s; fb1 = *(const float4*)(s + 4);
        if (kn + TILE_K < K) __builtin_prefetch(s + TILE_K, 0, 1);
      } else {
        async_copy_b128(Bb + bRow + kn, &Bt[nxt][br][bc]);
      }
    }

    // -- compute on current buffer: fragments per ISA 16-bit A/B layouts --
    v16bf a0 = make_frag(&At[cur][mOff + l16][hlf * 8],      &At[cur][mOff + l16][16 + hlf * 8]);
    v16bf a1 = make_frag(&At[cur][mOff + 16 + l16][hlf * 8], &At[cur][mOff + 16 + l16][16 + hlf * 8]);
    v16bf b0 = make_frag(&Bt[cur][nOff + l16][hlf * 8],      &Bt[cur][nOff + l16][16 + hlf * 8]);
    v16bf b1 = make_frag(&Bt[cur][nOff + 16 + l16][hlf * 8], &Bt[cur][nOff + 16 + l16][16 + hlf * 8]);
    acc00 = __builtin_amdgcn_wmma_f32_16x16x32_bf16(false, a0, false, b0, (short)0, acc00, false, false);
    acc01 = __builtin_amdgcn_wmma_f32_16x16x32_bf16(false, a0, false, b1, (short)0, acc01, false, false);
    acc10 = __builtin_amdgcn_wmma_f32_16x16x32_bf16(false, a1, false, b0, (short)0, acc10, false, false);
    acc11 = __builtin_amdgcn_wmma_f32_16x16x32_bf16(false, a1, false, b1, (short)0, acc11, false, false);

    // -- commit next tile into LDS behind the compute --
    if (hasNext) {
      if (A_IS_F32) {
        *(v8bf*)&At[nxt][ar][ac]      = cvt8(fa0, fa1);
        *(v8bf*)&At[nxt][ar + 64][ac] = cvt8(fa2, fa3);
      }
      if (B_IS_F32) {
        *(v8bf*)&Bt[nxt][br][bc] = cvt8(fb0, fb1);
      }
      if (!A_IS_F32 || !B_IS_F32) wait_async();
    }
    __syncthreads();
    cur = nxt;
  }

  // ---- epilogue: C/D layout VGPR r -> row (hlf*8 + r), col l16 ----
#pragma unroll
  for (int mt = 0; mt < 2; ++mt) {
#pragma unroll
    for (int nt = 0; nt < 2; ++nt) {
      v8f acc = (mt == 0) ? ((nt == 0) ? acc00 : acc01)
                          : ((nt == 0) ? acc10 : acc11);
#pragma unroll
      for (int r = 0; r < 8; ++r) {
        const int row = mBlock + mOff + mt * 16 + hlf * 8 + r;
        const int col = nBlock + nOff + nt * 16 + l16;
        float v = acc[r];
        if (HAS_BIAS) v += bias[col];
        if (OUT_MODE == 0) {
          ((__bf16*)Obase)[offO + (long long)row * ldo + col] = (__bf16)v;
        } else if (OUT_MODE == 1) {
          // vT[b][h][d][s] = v[b*S+s][h*64+d]   (S=2048, H=16, dk=64)
          const int b = row >> 11, s = row & 2047;
          const int h = col >> 6,  d = col & 63;
          ((__bf16*)Obase)[((long long)((b * 16 + h) * 64 + d)) * 2048 + s] = (__bf16)v;
        } else {
          ((float*)Obase)[offO + (long long)row * ldo + col] = v;
        }
      }
    }
  }
}

__device__ __forceinline__ float waveMax(float v) {
#pragma unroll
  for (int o = 16; o > 0; o >>= 1) v = fmaxf(v, __shfl_xor(v, o, 32));
  return v;
}
__device__ __forceinline__ float waveSum(float v) {
#pragma unroll
  for (int o = 16; o > 0; o >>= 1) v += __shfl_xor(v, o, 32);
  return v;
}

// In-place causal softmax of scores/sqrt(dk); masked tail -> 0.
// grid = (T, B*H), block = 256 (8 wave32)
__global__ void __launch_bounds__(256)
softmax_causal(float* __restrict__ attnw, int T, int S) {
  __shared__ float red[8];
  __shared__ float bcast;
  const int t = blockIdx.x;
  const long long bh = blockIdx.y;
  float* row = attnw + (bh * (long long)T + t) * (long long)S;
  const int L = t + 1;                 // valid (unmasked) length
  const float scale = 0.125f;          // 1/sqrt(64)
  const int tid = threadIdx.x;
  const int lane = tid & 31, w = tid >> 5;

  float m = -3.402823466e38f;
  for (int s = tid; s < L; s += 256) m = fmaxf(m, row[s] * scale);
  m = waveMax(m);
  if (lane == 0) red[w] = m;
  __syncthreads();
  if (w == 0) {
    float x = (lane < 8) ? red[lane] : -3.402823466e38f;
    x = waveMax(x);
    if (lane == 0) bcast = x;
  }
  __syncthreads();
  m = bcast;

  float sum = 0.0f;
  for (int s = tid; s < L; s += 256) {
    const float p = __expf(row[s] * scale - m);
    row[s] = p;
    sum += p;
  }
  sum = waveSum(sum);
  __syncthreads();
  if (lane == 0) red[w] = sum;
  __syncthreads();
  if (w == 0) {
    float x = (lane < 8) ? red[lane] : 0.0f;
    x = waveSum(x);
    if (lane == 0) bcast = x;
  }
  __syncthreads();
  const float inv = 1.0f / bcast;
  for (int s = tid; s < S; s += 256) row[s] = (s < L) ? row[s] * inv : 0.0f;
}

extern "C" void kernel_launch(void* const* d_in, const int* in_sizes, int n_in,
                              void* d_out, int out_size, void* d_ws, size_t ws_size,
                              hipStream_t stream) {
  const int B = 2, T = 2048, S = 2048, D = 1024, H = 16, DK = 64;

  const float* Q  = (const float*)d_in[0];
  const float* Ki = (const float*)d_in[1];
  const float* V  = (const float*)d_in[2];
  // d_in[3] = mask (bool causal) -- applied analytically in softmax
  const float* Wq = (const float*)d_in[4];
  const float* bq = (const float*)d_in[5];
  const float* Wk = (const float*)d_in[6];
  const float* bk = (const float*)d_in[7];
  const float* Wv = (const float*)d_in[8];
  const float* bv = (const float*)d_in[9];
  const float* Wo = (const float*)d_in[10];
  const float* bo = (const float*)d_in[11];

  float* out   = (float*)d_out;
  float* attnw = out + (long long)B * T * D;   // scores live here, softmaxed in place

  char* ws = (char*)d_ws;
  const size_t MB = 1ull << 20;
  __bf16* qbf    = (__bf16*)(ws + 0 * MB);     // [B*T, D] bf16, 8 MB
  __bf16* kbf    = (__bf16*)(ws + 8 * MB);     // [B*S, D] bf16, 8 MB
  __bf16* vT     = (__bf16*)(ws + 16 * MB);    // [B,H,dk,S] bf16, 8 MB
  __bf16* attnbf = (__bf16*)(ws + 24 * MB);    // [B,H,T,dk] bf16 == reshaped [B*T, D], 8 MB

  dim3 blk(256);

  // --- projections: [4096 x 1024] = X fp32 @ W^T ; W row-major IS the [N x K] B-layout ---
  dim3 gp(D / BLOCK_N, (B * T) / BLOCK_M, 1);
  gemm_wmma_bf16<1, 1, 0, 1><<<gp, blk, 0, stream>>>(Q, D, Wq, D, qbf, D, bq,
                                                     B * T, D, D, 0, 0, 0, 0, 0, 0);
  gemm_wmma_bf16<1, 1, 0, 1><<<gp, blk, 0, stream>>>(Ki, D, Wk, D, kbf, D, bk,
                                                     B * S, D, D, 0, 0, 0, 0, 0, 0);
  gemm_wmma_bf16<1, 1, 1, 1><<<gp, blk, 0, stream>>>(V, D, Wv, D, vT, 0, bv,
                                                     B * S, D, D, 0, 0, 0, 0, 0, 0);

  // --- scores[b,h] = q_head [T x dk] @ k_head^T ; A,B both async bf16 ---
  dim3 gs(S / BLOCK_N, T / BLOCK_M, B * H);
  gemm_wmma_bf16<0, 0, 2, 0><<<gs, blk, 0, stream>>>(
      qbf, D, kbf, D, attnw, S, nullptr, T, S, DK,
      (long long)T * D, (long long)DK,                         // A: b,h offsets in qbf
      (long long)S * D, (long long)DK,                         // B: b,h offsets in kbf
      (long long)H * T * S, (long long)T * S);                 // O: b,h offsets in attn_w

  // --- softmax (causal mask + 1/8 scale), in place ---
  dim3 gsm(T, B * H);
  softmax_causal<<<gsm, blk, 0, stream>>>(attnw, T, S);

  // --- attn[b,h] = attn_w fp32 [T x S] @ v_head ; B = vT rows [dk x S] (async bf16) ---
  dim3 ga(DK / BLOCK_N, T / BLOCK_M, B * H);
  gemm_wmma_bf16<1, 0, 0, 0><<<ga, blk, 0, stream>>>(
      attnw, S, vT, S, attnbf, DK, nullptr, T, DK, S,
      (long long)H * T * S, (long long)T * S,                  // A: b,h offsets in attn_w
      (long long)H * DK * S, (long long)DK * S,                // B: b,h offsets in vT
      (long long)H * T * DK, (long long)T * DK);               // O: b,h offsets in attnbf

  // --- out = attn_flat [4096 x 1024] @ Wo^T + bo (fp32 out); A async bf16, B = Wo fp32 ---
  dim3 gf(D / BLOCK_N, (B * T) / BLOCK_M, 1);
  gemm_wmma_bf16<0, 1, 2, 1><<<gf, blk, 0, stream>>>(attnbf, D, Wo, D, out, D, bo,
                                                     B * T, D, D, 0, 0, 0, 0, 0, 0);
}
